// SimCore_13245679141527
// MI455X (gfx1250) — matrix operand order repaired and checked by reading, not compiled, not run on hardware
//
#include <hip/hip_runtime.h>
#include <hip/hip_bf16.h>
#include <math.h>

// ---------------------------------------------------------------------------
// Types for CDNA5 WMMA (bf16 16x16x32, f32 accum)
// ---------------------------------------------------------------------------
typedef __attribute__((ext_vector_type(16))) __bf16 v16bf;
typedef __attribute__((ext_vector_type(8)))  __bf16 v8bf;
typedef __attribute__((ext_vector_type(8)))  float  v8f;
typedef __bf16 bf16_t;

#define NT   6
#define NG   4
#define BB   256
#define HH   1024
#define AA   64
#define TT   128
#define LO_  30
#define HEADW 407
#define LOGW  814

__device__ __forceinline__ v8f wmma_bf16(v16bf a, v16bf b, v8f c) {
  return __builtin_amdgcn_wmma_f32_16x16x32_bf16(
      /*neg_a=*/false, a, /*neg_b=*/false, b,
      /*c_mod=*/(short)0, c, /*reuse_a=*/false, /*reuse_b=*/false);
}

// Concatenate two 16-byte halves into one fragment: pure register coalesce.
__device__ __forceinline__ v16bf make_frag(v8bf lo, v8bf hi) {
  union { v8bf h[2]; v16bf v; } u;
  u.h[0] = lo; u.h[1] = hi;
  return u.v;
}

// A fragment (16x32 bf16, row-major source, rows = M):
//  lane m (lo half) holds row m, K = {k0..k0+15}
//  lane m (hi half) holds row m, K = {k0+16..k0+31}
__device__ __forceinline__ v16bf load_a_frag(const bf16_t* __restrict__ base,
                                             int row0, int stride, int k0, int lane) {
  const int m = lane & 15, half = lane >> 4;
  const bf16_t* p = base + (long)(row0 + m) * stride + k0 + half * 16;
  return make_frag(*(const v8bf*)p, *(const v8bf*)(p + 8));
}

// B fragment for C = A * W^T, W stored row-major (N x K):
//  lane n (lo half) holds W[n0+n][k0..k0+15]  (== column n of B, K 0..15)
//  lane n (hi half) holds W[n0+n][k0+16..k0+31]
__device__ __forceinline__ v16bf load_b_frag(const bf16_t* __restrict__ wbase,
                                             int n0, int stride, int k0, int lane) {
  const int n = lane & 15, half = lane >> 4;
  const bf16_t* p = wbase + (long)(n0 + n) * stride + k0 + half * 16;
  return make_frag(*(const v8bf*)p, *(const v8bf*)(p + 8));
}

__device__ __forceinline__ v16bf load_b_frag_clamp(const bf16_t* __restrict__ wbase,
                                                   int n0, int stride, int k0,
                                                   int lane, int nmax) {
  int n = n0 + (lane & 15);
  n = n < nmax ? n : (nmax - 1);
  const int half = lane >> 4;
  const bf16_t* p = wbase + (long)n * stride + k0 + half * 16;
  return make_frag(*(const v8bf*)p, *(const v8bf*)(p + 8));
}

// LDS fragment assembly (padded tile, row stride PSTR elements)
__device__ __forceinline__ v16bf lds_a_frag(const bf16_t* p) {
  return make_frag(*(const v8bf*)p, *(const v8bf*)(p + 8));
}
__device__ __forceinline__ v16bf lds_b_frag(const bf16_t* p) {
  return make_frag(*(const v8bf*)p, *(const v8bf*)(p + 8));
}

// ---------------------------------------------------------------------------
// CDNA5 async global->LDS copy (ASYNCcnt-tracked DMA; no VGPR round-trip).
// GVS addressing: global address = 64-bit SGPR base + per-lane 32-bit VGPR byte offset + immediate offset;
// LDS destination = per-lane VGPR LDS address + the same immediate offset (applies to both sides).
// ---------------------------------------------------------------------------
__device__ __forceinline__ void async_ld128(unsigned ldsoff, unsigned voff,
                                            const void* sbase) {
  asm volatile("global_load_async_to_lds_b128 %0, %1, %2"
               :: "v"(ldsoff), "v"(voff), "s"(sbase) : "memory");
}
__device__ __forceinline__ void async_ld128_o16(unsigned ldsoff, unsigned voff,
                                                const void* sbase) {
  asm volatile("global_load_async_to_lds_b128 %0, %1, %2 offset:16"
               :: "v"(ldsoff), "v"(voff), "s"(sbase) : "memory");
}
__device__ __forceinline__ void wait_asynccnt0() {
  asm volatile("s_wait_asynccnt 0" ::: "memory");
}

__device__ __forceinline__ float sigmoidf_(float x) { return 1.0f / (1.0f + expf(-x)); }
__device__ __forceinline__ float softplusf_(float x) {
  return fmaxf(x, 0.0f) + log1pf(expf(-fabsf(x)));
}

// ---------------------------------------------------------------------------
// Elementwise f32 -> bf16 convert
// ---------------------------------------------------------------------------
__global__ void f2bf_kernel(const float* __restrict__ src, bf16_t* __restrict__ dst, long n) {
  long i = (long)blockIdx.x * blockDim.x + threadIdx.x;
  if (i < n) dst[i] = (bf16_t)src[i];
}

// Gather h0 = beliefs[idx_i[t]] into state buffers (fp32 + bf16 mirror)
__global__ void init_h_kernel(const float* __restrict__ beliefs,
                              const int* __restrict__ idx_i,
                              float* __restrict__ h0, bf16_t* __restrict__ h0bf) {
  long i = (long)blockIdx.x * blockDim.x + threadIdx.x;
  const long per_t = (long)BB * HH;
  if (i >= (long)NT * per_t) return;
  int t = (int)(i / per_t);
  long rest = i - (long)t * per_t;
  float v = beliefs[(long)idx_i[t] * per_t + rest];
  h0[i] = v;
  h0bf[i] = (bf16_t)v;
}

// ---------------------------------------------------------------------------
// One GRU step for all 6 trajectories, LDS double-buffered via async DMA.
//   gates = a @ W_ih^T + h @ W_hh^T (+ biases in epilogue)
//   r,z combined; n-gate keeps input/hidden parts separate (n = tanh(in + r*hn))
// grid: (H/64, B/64, NT) = (16,4,6); block 256 (8 waves).
// Block tile: 64 batch rows x 64 cols. Wave = 2 M-subtiles x 16 cols.
// Stage = 64 K values: 12 WMMA per barrier, 8 f32 accumulators per wave.
// Pipeline: async global->LDS DMA of stage c+1 overlaps WMMA on stage c;
// s_wait_asynccnt 0 + barrier after compute.
// ---------------------------------------------------------------------------
#define CH   64   // staged K-chunk
#define PSTR 72   // padded LDS row stride (bf16 elements); 144B, 16B-aligned segs

__global__ __launch_bounds__(256) void gru_step_kernel(
    const float*  __restrict__ h_cur, const bf16_t* __restrict__ h_cur_bf,
    float*        __restrict__ h_next, bf16_t*      __restrict__ h_next_bf,
    const bf16_t* __restrict__ act_bf, const bf16_t* __restrict__ w_ih_bf,
    const bf16_t* __restrict__ w_hh_bf, const float* __restrict__ b_ih,
    const float*  __restrict__ b_hh, const int* __restrict__ idx_i,
    const int*    __restrict__ Kmat, float* __restrict__ saved,
    bf16_t*       __restrict__ saved_bf, int s) {
  __shared__ bf16_t sA[2][64 * PSTR];       // h tile: 64 rows x 64 K
  __shared__ bf16_t sB[2][3][64 * PSTR];    // w_hh tiles: 3 gates x 64 N x 64 K

  const int t    = blockIdx.z;
  const int b0   = blockIdx.y * 64;
  const int n0   = blockIdx.x * 64;
  const int tid  = threadIdx.x;
  const int wave = tid >> 5;
  const int lane = tid & 31;
  const int half = lane >> 4;
  const int l16  = lane & 15;
  const int rowg = wave & 1;        // 0,1 -> 32-row group
  const int colg = wave >> 1;       // 0..3 -> 16-col group
  const int mbase = b0 + rowg * 32;
  const int ncolw = n0 + colg * 16;

  const long hbase = (long)t * BB * HH;
  v8f acc_r0 = {}, acc_r1 = {}, acc_z0 = {}, acc_z1 = {};
  v8f acc_in0 = {}, acc_in1 = {}, acc_hn0 = {}, acc_hn1 = {};

  // ---- input part: a @ W_ih^T  (K = 64, direct global) ----
  const int i0 = idx_i[t];
  const bf16_t* abase = act_bf + (long)(i0 + s) * BB * AA;
#pragma unroll
  for (int k0 = 0; k0 < AA; k0 += 32) {
    v16bf af0 = load_a_frag(abase, mbase,      AA, k0, lane);
    v16bf af1 = load_a_frag(abase, mbase + 16, AA, k0, lane);
    v16bf br = load_b_frag(w_ih_bf,                     ncolw, AA, k0, lane);
    v16bf bz = load_b_frag(w_ih_bf + (long)HH * AA,     ncolw, AA, k0, lane);
    v16bf bn = load_b_frag(w_ih_bf + (long)2 * HH * AA, ncolw, AA, k0, lane);
    acc_r0  = wmma_bf16(af0, br, acc_r0);   acc_r1  = wmma_bf16(af1, br, acc_r1);
    acc_z0  = wmma_bf16(af0, bz, acc_z0);   acc_z1  = wmma_bf16(af1, bz, acc_z1);
    acc_in0 = wmma_bf16(af0, bn, acc_in0);  acc_in1 = wmma_bf16(af1, bn, acc_in1);
  }

  // ---- hidden part: h @ W_hh^T  (K = 1024), async-DMA double-buffered ----
  const bf16_t* hb = h_cur_bf + hbase;
  const int frow = tid >> 2;            // 0..63 : row this thread stages
  const int fcol = (tid & 3) * 16;      // 0,16,32,48 : K-segment (16 bf16 = 32B)
  const int sOfs = frow * PSTR + fcol;

  // per-lane byte offsets (VADDR) into each source, fixed across stages
  const unsigned vA  = (unsigned)(((b0 + frow) * HH + fcol) * 2);
  const unsigned vB0 = (unsigned)(((n0 + frow) * HH + fcol) * 2);
  const unsigned vB1 = (unsigned)(((HH + n0 + frow) * HH + fcol) * 2);
  const unsigned vB2 = (unsigned)(((2 * HH + n0 + frow) * HH + fcol) * 2);
  // per-lane LDS byte offsets (VDST) for both buffers
  const unsigned dA0  = (unsigned)(size_t)&sA[0][sOfs];
  const unsigned dA1  = (unsigned)(size_t)&sA[1][sOfs];
  const unsigned dB00 = (unsigned)(size_t)&sB[0][0][sOfs];
  const unsigned dB01 = (unsigned)(size_t)&sB[1][0][sOfs];
  const unsigned dB10 = (unsigned)(size_t)&sB[0][1][sOfs];
  const unsigned dB11 = (unsigned)(size_t)&sB[1][1][sOfs];
  const unsigned dB20 = (unsigned)(size_t)&sB[0][2][sOfs];
  const unsigned dB21 = (unsigned)(size_t)&sB[1][2][sOfs];

  // prologue: DMA stage 0 into buffer 0
  async_ld128    (dA0,  vA,  hb);        async_ld128_o16(dA0,  vA,  hb);
  async_ld128    (dB00, vB0, w_hh_bf);   async_ld128_o16(dB00, vB0, w_hh_bf);
  async_ld128    (dB10, vB1, w_hh_bf);   async_ld128_o16(dB10, vB1, w_hh_bf);
  async_ld128    (dB20, vB2, w_hh_bf);   async_ld128_o16(dB20, vB2, w_hh_bf);
  wait_asynccnt0();
  __syncthreads();

  const int aOfs = (rowg * 32 + l16) * PSTR + half * 16;
  const int bOfs = (colg * 16 + l16) * PSTR + half * 16;

  for (int c = 0; c < HH / CH; ++c) {
    const int buf = c & 1;
    if (c + 1 < HH / CH) {  // DMA next stage into the other buffer
      const int k0 = (c + 1) * CH;
      const bf16_t* hs = hb + k0;
      const bf16_t* wsrc = w_hh_bf + k0;
      const unsigned da  = buf ? dA0  : dA1;
      const unsigned db0 = buf ? dB00 : dB01;
      const unsigned db1 = buf ? dB10 : dB11;
      const unsigned db2 = buf ? dB20 : dB21;
      async_ld128    (da,  vA,  hs);    async_ld128_o16(da,  vA,  hs);
      async_ld128    (db0, vB0, wsrc);  async_ld128_o16(db0, vB0, wsrc);
      async_ld128    (db1, vB1, wsrc);  async_ld128_o16(db1, vB1, wsrc);
      async_ld128    (db2, vB2, wsrc);  async_ld128_o16(db2, vB2, wsrc);
    }
    // compute from LDS (overlaps with the async DMA above)
    const bf16_t* sa = &sA[buf][0];
#pragma unroll
    for (int ksub = 0; ksub < CH; ksub += 32) {
      v16bf af0 = lds_a_frag(sa + aOfs + ksub);
      v16bf af1 = lds_a_frag(sa + aOfs + ksub + 16 * PSTR);
      v16bf br  = lds_b_frag(&sB[buf][0][bOfs + ksub]);
      v16bf bz  = lds_b_frag(&sB[buf][1][bOfs + ksub]);
      v16bf bn  = lds_b_frag(&sB[buf][2][bOfs + ksub]);
      acc_r0  = wmma_bf16(af0, br, acc_r0);   acc_r1  = wmma_bf16(af1, br, acc_r1);
      acc_z0  = wmma_bf16(af0, bz, acc_z0);   acc_z1  = wmma_bf16(af1, bz, acc_z1);
      acc_hn0 = wmma_bf16(af0, bn, acc_hn0);  acc_hn1 = wmma_bf16(af1, bn, acc_hn1);
    }
    wait_asynccnt0();   // this wave's DMA for stage c+1 landed in LDS
    __syncthreads();    // everyone's DMA landed
  }

  // ---- epilogue: gate nonlinearities + state update ----
  const int col = ncolw + l16;
  const float br_  = b_ih[col]        + b_hh[col];
  const float bz_  = b_ih[col + 1024] + b_hh[col + 1024];
  const float bin_ = b_ih[col + 2048];
  const float bhn_ = b_hh[col + 2048];
  int kg0 = Kmat[t * 4 + 0], kg1 = Kmat[t * 4 + 1];
  int kg2 = Kmat[t * 4 + 2], kg3 = Kmat[t * 4 + 3];

#pragma unroll
  for (int st = 0; st < 2; ++st) {
    const v8f& ar = st ? acc_r1 : acc_r0;
    const v8f& az = st ? acc_z1 : acc_z0;
    const v8f& ai = st ? acc_in1 : acc_in0;
    const v8f& ah = st ? acc_hn1 : acc_hn0;
    const int rbase = mbase + st * 16 + half * 8;
#pragma unroll
    for (int v = 0; v < 8; ++v) {
      const int row = rbase + v;
      const long ofs = hbase + (long)row * HH + col;
      float hold = h_cur[ofs];
      float r = sigmoidf_(ar[v] + br_);
      float z = sigmoidf_(az[v] + bz_);
      float n = tanhf(ai[v] + bin_ + r * (ah[v] + bhn_));
      float hnew = (1.0f - z) * n + z * hold;
      h_next[ofs]    = hnew;
      h_next_bf[ofs] = (bf16_t)hnew;
#pragma unroll
      for (int g = 0; g < 4; ++g) {
        int kg = (g == 0) ? kg0 : (g == 1) ? kg1 : (g == 2) ? kg2 : kg3;
        if (kg == s) {
          long so = ((long)(t * NG + g) * BB + row) * HH + col;
          saved[so]    = hnew;
          saved_bf[so] = (bf16_t)hnew;
        }
      }
    }
  }
}

// ---------------------------------------------------------------------------
// MLP1: h1 = relu(saved @ pre_w1^T + b1)  — M=6144, N=64, K=1024
// ---------------------------------------------------------------------------
__global__ __launch_bounds__(256) void mlp1_kernel(
    const bf16_t* __restrict__ saved_bf, const bf16_t* __restrict__ w1_bf,
    const float* __restrict__ b1, bf16_t* __restrict__ h1_bf) {
  const int m0   = blockIdx.y * 64;
  const int n0   = blockIdx.x * 32;
  const int wave = threadIdx.x >> 5, lane = threadIdx.x & 31;
  const int mrow = m0 + ((wave & 3) << 4);
  const int ncol = n0 + ((wave >> 2) << 4);
  v8f acc = {};
  for (int k0 = 0; k0 < HH; k0 += 32) {
    v16bf a = load_a_frag(saved_bf, mrow, HH, k0, lane);
    v16bf b = load_b_frag(w1_bf, ncol, HH, k0, lane);
    acc = wmma_bf16(a, b, acc);
  }
  const int col = ncol + (lane & 15);
  const int rbase = mrow + ((lane >> 4) << 3);
  const float bias = b1[col];
#pragma unroll
  for (int v = 0; v < 8; ++v) {
    float x = fmaxf(acc[v] + bias, 0.0f);
    h1_bf[(long)(rbase + v) * 64 + col] = (bf16_t)x;
  }
}

// ---------------------------------------------------------------------------
// MLP2: x = saved + relu(h1 @ pre_w2^T + b2)  — M=6144, N=1024, K=64
// ---------------------------------------------------------------------------
__global__ __launch_bounds__(256) void mlp2_kernel(
    const bf16_t* __restrict__ h1_bf, const bf16_t* __restrict__ w2_bf,
    const float* __restrict__ b2, const float* __restrict__ saved,
    bf16_t* __restrict__ x_bf) {
  const int m0   = blockIdx.y * 64;
  const int n0   = blockIdx.x * 32;
  const int wave = threadIdx.x >> 5, lane = threadIdx.x & 31;
  const int mrow = m0 + ((wave & 3) << 4);
  const int ncol = n0 + ((wave >> 2) << 4);
  v8f acc = {};
#pragma unroll
  for (int k0 = 0; k0 < 64; k0 += 32) {
    v16bf a = load_a_frag(h1_bf, mrow, 64, k0, lane);
    v16bf b = load_b_frag(w2_bf, ncol, 64, k0, lane);
    acc = wmma_bf16(a, b, acc);
  }
  const int col = ncol + (lane & 15);
  const int rbase = mrow + ((lane >> 4) << 3);
  const float bias = b2[col];
#pragma unroll
  for (int v = 0; v < 8; ++v) {
    long ofs = (long)(rbase + v) * HH + col;
    float x = saved[ofs] + fmaxf(acc[v] + bias, 0.0f);
    x_bf[ofs] = (bf16_t)x;
  }
}

// ---------------------------------------------------------------------------
// Decoder: logits = x @ dec_w^T + dec_b  — M=6144, N=814 (tiled to 832), K=1024
// ---------------------------------------------------------------------------
__global__ __launch_bounds__(256) void dec_kernel(
    const bf16_t* __restrict__ x_bf, const bf16_t* __restrict__ dw_bf,
    const float* __restrict__ db, float* __restrict__ logits) {
  const int m0   = blockIdx.y * 64;
  const int n0   = blockIdx.x * 32;
  const int wave = threadIdx.x >> 5, lane = threadIdx.x & 31;
  const int mrow = m0 + ((wave & 3) << 4);
  const int ncol = n0 + ((wave >> 2) << 4);
  v8f acc = {};
  for (int k0 = 0; k0 < HH; k0 += 32) {
    v16bf a = load_a_frag(x_bf, mrow, HH, k0, lane);
    v16bf b = load_b_frag_clamp(dw_bf, ncol, HH, k0, lane, LOGW);
    acc = wmma_bf16(a, b, acc);
  }
  const int col = ncol + (lane & 15);
  const int rbase = mrow + ((lane >> 4) << 3);
  if (col < LOGW) {
    const float bias = db[col];
#pragma unroll
    for (int v = 0; v < 8; ++v)
      logits[(long)(rbase + v) * LOGW + col] = acc[v] + bias;
  }
}

// ---------------------------------------------------------------------------
// Loss: one wave per sample (6144 samples), both players; atomicAdd scalar.
// ---------------------------------------------------------------------------
__global__ __launch_bounds__(256) void loss_kernel(
    const float* __restrict__ logits, const float* __restrict__ obs,
    const int* __restrict__ idx_i, const int* __restrict__ Kmat,
    float* __restrict__ out) {
  const int wave = threadIdx.x >> 5, lane = threadIdx.x & 31;
  const int sid = blockIdx.x * 8 + wave;  // grid sized so sid < 6144
  const int t = sid / (NG * BB);
  const int g = (sid / BB) % NG;
  const int b = sid % BB;
  const int k = Kmat[t * NG + g];
  const int i0 = idx_i[t];
  const float* lg = logits + (long)sid * LOGW;
  const float* ob = obs + ((long)(k + i0 + 1) * BB + b) * 16;

  float contrib = 0.0f;
#pragma unroll
  for (int p = 0; p < 2; ++p) {
    const float* l = lg + p * HEADW;
    const float* tg = ob + p * 8;
    float mx = -3.4e38f;
    for (int j = lane; j < 400; j += 32) mx = fmaxf(mx, l[j]);
#pragma unroll
    for (int o = 16; o >= 1; o >>= 1) mx = fmaxf(mx, __shfl_xor(mx, o, 32));
    float sum = 0.0f;
    for (int j = lane; j < 400; j += 32) sum += expf(l[j] - mx);
#pragma unroll
    for (int o = 16; o >= 1; o >>= 1) sum += __shfl_xor(sum, o, 32);
    if (lane == 0) {
      int cls = (int)tg[0];
      cls = cls < 0 ? 0 : (cls > 399 ? 399 : cls);
      float ce = -(l[cls] - mx - logf(sum));
      float d0 = l[400] - tg[1];
      float d1 = l[401] - tg[2];
      float s2 = softplusf_(l[402]) - tg[3];
      float s3 = softplusf_(l[403]) - tg[4];
      float bc = (softplusf_(l[404]) - l[404] * tg[5]) +
                 (softplusf_(l[405]) - l[405] * tg[6]) +
                 (softplusf_(l[406]) - l[406] * tg[7]);
      contrib += ce * (1.0f / 1024.0f) + (d0 * d0 + d1 * d1) * (1.0f / 2048.0f) +
                 (s2 * s2 + s3 * s3 + bc) * (1.0f / 1024.0f);
    }
  }
  if (lane == 0) atomicAdd(out, contrib);
}

__global__ void zero_out_kernel(float* out) {
  if (threadIdx.x == 0 && blockIdx.x == 0) out[0] = 0.0f;
}

// ---------------------------------------------------------------------------
// Launch
// ---------------------------------------------------------------------------
extern "C" void kernel_launch(void* const* d_in, const int* in_sizes, int n_in,
                              void* d_out, int out_size, void* d_ws, size_t ws_size,
                              hipStream_t stream) {
  const float* observations = (const float*)d_in[0];
  const float* actions      = (const float*)d_in[1];
  const float* beliefs      = (const float*)d_in[2];
  const float* w_ih         = (const float*)d_in[3];
  const float* w_hh         = (const float*)d_in[4];
  const float* b_ih         = (const float*)d_in[5];
  const float* b_hh         = (const float*)d_in[6];
  const float* pre_w1       = (const float*)d_in[7];
  const float* pre_b1       = (const float*)d_in[8];
  const float* pre_w2       = (const float*)d_in[9];
  const float* pre_b2       = (const float*)d_in[10];
  const float* dec_w        = (const float*)d_in[11];
  const float* dec_b        = (const float*)d_in[12];
  const int*   idx_i        = (const int*)d_in[13];
  const int*   Kmat         = (const int*)d_in[14];
  float* out = (float*)d_out;

  // ---- workspace carve-up (64B aligned slabs) ----
  char* ws = (char*)d_ws;
  size_t off = 0;
  auto alloc = [&](size_t bytes) {
    void* p = ws + off;
    off += (bytes + 63) & ~size_t(63);
    return p;
  };
  const long NtBH = (long)NT * BB * HH;        // 1,572,864
  const long MROW = (long)NT * NG * BB;        // 6144

  float*  h_f32[2];  bf16_t* h_bf[2];
  h_f32[0] = (float*)alloc(NtBH * 4);
  h_f32[1] = (float*)alloc(NtBH * 4);
  h_bf[0]  = (bf16_t*)alloc(NtBH * 2);
  h_bf[1]  = (bf16_t*)alloc(NtBH * 2);
  float*  saved    = (float*)alloc((long)NG * NtBH * 4);
  bf16_t* saved_bf = (bf16_t*)alloc((long)NG * NtBH * 2);
  bf16_t* w_ih_bf  = (bf16_t*)alloc((long)3 * HH * AA * 2);
  bf16_t* w_hh_bf  = (bf16_t*)alloc((long)3 * HH * HH * 2);
  bf16_t* w1_bf    = (bf16_t*)alloc((long)64 * HH * 2);
  bf16_t* w2_bf    = (bf16_t*)alloc((long)HH * 64 * 2);
  bf16_t* dw_bf    = (bf16_t*)alloc((long)LOGW * HH * 2);
  bf16_t* act_bf   = (bf16_t*)alloc((long)TT * BB * AA * 2);
  bf16_t* h1_bf    = (bf16_t*)alloc(MROW * 64 * 2);
  bf16_t* x_bf     = (bf16_t*)alloc(MROW * HH * 2);
  float*  logits   = (float*)alloc(MROW * LOGW * 4);

  // ---- one-time f32 -> bf16 conversions ----
  auto cvt = [&](const float* s, bf16_t* d, long n) {
    f2bf_kernel<<<(unsigned)((n + 255) / 256), 256, 0, stream>>>(s, d, n);
  };
  cvt(w_ih,   w_ih_bf, (long)3 * HH * AA);
  cvt(w_hh,   w_hh_bf, (long)3 * HH * HH);
  cvt(pre_w1, w1_bf,   (long)64 * HH);
  cvt(pre_w2, w2_bf,   (long)HH * 64);
  cvt(dec_w,  dw_bf,   (long)LOGW * HH);
  cvt(actions, act_bf, (long)TT * BB * AA);

  // ---- h0 = beliefs[idx_i] ----
  init_h_kernel<<<(unsigned)((NtBH + 255) / 256), 256, 0, stream>>>(
      beliefs, idx_i, h_f32[0], h_bf[0]);

  // ---- 30 sequential GRU steps ----
  dim3 sgrid(HH / 64, BB / 64, NT);  // (16, 4, 6)
  for (int s = 0; s < LO_; ++s) {
    int cur = s & 1, nxt = cur ^ 1;
    gru_step_kernel<<<sgrid, 256, 0, stream>>>(
        h_f32[cur], h_bf[cur], h_f32[nxt], h_bf[nxt],
        act_bf, w_ih_bf, w_hh_bf, b_ih, b_hh, idx_i, Kmat,
        saved, saved_bf, s);
  }

  // ---- residual MLP + decoder ----
  mlp1_kernel<<<dim3(2,  (unsigned)(MROW / 64)), 256, 0, stream>>>(saved_bf, w1_bf, pre_b1, h1_bf);
  mlp2_kernel<<<dim3(32, (unsigned)(MROW / 64)), 256, 0, stream>>>(h1_bf, w2_bf, pre_b2, saved, x_bf);
  dec_kernel <<<dim3(26, (unsigned)(MROW / 64)), 256, 0, stream>>>(x_bf, dw_bf, dec_b, logits);

  // ---- loss reduction ----
  zero_out_kernel<<<1, 32, 0, stream>>>(out);
  loss_kernel<<<(unsigned)(MROW / 8), 256, 0, stream>>>(logits, observations, idx_i, Kmat, out);
}
